// FUB_36344013259307
// MI455X (gfx1250) — matrix operand order repaired and checked by reading, not compiled, not run on hardware
//
#include <hip/hip_runtime.h>
#include <hip/hip_bf16.h>
#include <stdint.h>

typedef __attribute__((ext_vector_type(16))) _Float16 v16h;
typedef __attribute__((ext_vector_type(8)))  _Float16 v8h;
typedef __attribute__((ext_vector_type(4)))  _Float16 h4;
typedef __attribute__((ext_vector_type(8)))  float    v8f;
typedef __attribute__((ext_vector_type(4)))  float    f4;
typedef __attribute__((ext_vector_type(4)))  unsigned int u32x4;
typedef __attribute__((ext_vector_type(8)))  int      i32x8;
typedef __attribute__((ext_vector_type(4)))  int      i32x4;

#define NN 5
#define BB 8
#define CC 256
#define HW 1024
#define THRESH 0.3f

#define MT 64            // pixels per block tile (M dimension)
#define NT 64            // output channels per block tile (N dimension)
#define KSTEP 32
#define LDS_STRIDE 264   // halfs per row: 256 + 8 pad -> rows 16B aligned, conflict-free
#define TILE_J (MT * LDS_STRIDE)          // halfs per x_j tile
#define WTILE  (NT * LDS_STRIDE)          // halfs per weight tile buffer
#define XTILES_BYTES (NN * TILE_J * 2)    // byte offset of weight double-buffer in LDS

// ---- workspace layout (bytes) ----
#define WS_AH    0
#define WS_W2H   (WS_AH + (size_t)NN*CC*CC*2)
#define WS_XH    (WS_W2H + (size_t)NN*CC*CC*2)
#define WS_CBUF  (WS_XH + (size_t)NN*BB*CC*HW*2)

__device__ __forceinline__ v16h combine16(v8h a0, v8h a1) {
  return __builtin_shufflevector(a0, a1, 0,1,2,3,4,5,6,7,8,9,10,11,12,13,14,15);
}

// TDM: DMA one 64-row x 256-half weight tile (row-major, stride 256) into LDS,
// inserting 4 DWORDs of padding after every 128 DWORDs (one row) so the LDS
// image has row stride 264 halfs. D# built per CDNA5 ISA sec 8.3/8.4.
__device__ __forceinline__ void tdm_load_weights(const _Float16* gsrc, uint32_t ldsByteOff) {
  uint64_t ga = (uint64_t)(uintptr_t)gsrc;
  u32x4 g0;
  g0[0] = 1u;                                               // count=1, user D#
  g0[1] = ldsByteOff;                                       // lds_addr (bytes)
  g0[2] = (uint32_t)(ga & 0xffffffffu);                     // global_addr[31:0]
  g0[3] = (uint32_t)((ga >> 32) & 0x01ffffffu) | (2u << 30);// global_addr[56:32] | type=2
  i32x8 g1;
  // dw0: wg_mask=0 | data_size=1(2B)<<16 | pad_enable<<20 | pad_interval=6(128dw)<<22 | pad_amount=3(4dw)<<25
  g1[0] = (1 << 16) | (1 << 20) | (6 << 22) | (3 << 25);
  // dw1: atomic_barrier_addr=0 ; tensor_dim0[15:0] (=256) in [31:16]
  g1[1] = (int)(256u << 16);
  // dw2: tensor_dim0[31:16]=0 in [15:0] ; tensor_dim1[15:0] (=64) in [31:16]
  g1[2] = (int)(64u << 16);
  // dw3: tensor_dim1[31:16]=0 ; tile_dim0 (=256) in [31:16]
  g1[3] = (int)(256u << 16);
  // dw4: tile_dim1 (=64) in [15:0] ; tile_dim2=0
  g1[4] = 64;
  // dw5: tensor_dim0_stride[31:0] = 256
  g1[5] = 256;
  g1[6] = 0;                                                // dim0_stride[47:32], dim1_stride lo
  g1[7] = 0;                                                // dim1_stride hi (unused, 2D)
  i32x4 z4 = {0, 0, 0, 0};
  i32x8 z8 = {0, 0, 0, 0, 0, 0, 0, 0};
  __builtin_amdgcn_tensor_load_to_lds(g0, g1, z4, z4, z8, 0);
}

// Ah[i][o][c] = half(W1+W2), W2h[i][o][c] = half(W2).  conv_w is [N][C][2C].
__global__ void prep_weights(const float* __restrict__ w,
                             _Float16* __restrict__ Ah, _Float16* __restrict__ W2h) {
  int idx = blockIdx.x * 256 + threadIdx.x;             // over N*C*C
  if (idx >= NN * CC * CC) return;
  int i = idx / (CC * CC);
  int rem = idx - i * CC * CC;
  int o = rem >> 8;
  int c = rem & 255;
  const float* wr = w + ((size_t)(i * CC + o)) * (2 * CC);
  float w1 = wr[c], w2 = wr[CC + c];
  Ah[idx]  = (_Float16)(w1 + w2);
  W2h[idx] = (_Float16)w2;
}

// x (fp32) -> xh (fp16), streaming, non-temporal.
__global__ void prep_x(const float* __restrict__ x, _Float16* __restrict__ xh) {
  size_t idx = ((size_t)blockIdx.x * 256 + threadIdx.x) * 4;
  f4 v = __builtin_nontemporal_load((const f4*)(x + idx));
  h4 h;
  h[0] = (_Float16)v[0]; h[1] = (_Float16)v[1];
  h[2] = (_Float16)v[2]; h[3] = (_Float16)v[3];
  __builtin_nontemporal_store(h, (h4*)(xh + idx));
}

// Stage a 64-pixel x 256-channel x-tile transposed into LDS: xT[pix][c],
// two channels per thread so LDS writes are 32-bit.
__device__ __forceinline__ void stage_xT(const _Float16* __restrict__ xbase,
                                         _Float16* __restrict__ xT, int tid) {
  for (int it = tid; it < (CC / 2) * (MT / 2); it += 256) {
    int cp = it >> 5, p2 = it & 31;            // cp: channel pair, p2: pixel pair
    int c0 = 2 * cp;
    uint32_t v0 = ((const uint32_t*)(xbase + (size_t)c0 * HW))[p2];
    uint32_t v1 = ((const uint32_t*)(xbase + (size_t)(c0 + 1) * HW))[p2];
    uint32_t lo = (v0 & 0xffffu) | (v1 << 16);          // pix 2*p2   : c0, c0+1
    uint32_t hi = (v0 >> 16) | (v1 & 0xffff0000u);      // pix 2*p2+1 : c0, c0+1
    *(uint32_t*)(xT + (2 * p2) * LDS_STRIDE + c0)     = lo;
    *(uint32_t*)(xT + (2 * p2 + 1) * LDS_STRIDE + c0) = hi;
  }
}

// cbuf[i][b][o][hw] = sum_c W2[i][o][c]*x[i][b][c][hw] + bias[i][o]   (f16 WMMA, f32 accum)
__global__ void ci_gemm(const _Float16* __restrict__ W2h, const _Float16* __restrict__ xh,
                        const float* __restrict__ cbias, float* __restrict__ cbuf) {
  __shared__ _Float16 xT[MT * LDS_STRIDE];
  const int tid = threadIdx.x;
  const int bI = blockIdx.y;
  const int i = blockIdx.z;
  const int pix0 = (blockIdx.x & 15) * MT;
  const int ch0  = (blockIdx.x >> 4) * NT;

  stage_xT(xh + ((size_t)(i * BB + bI) * CC) * HW + pix0, xT, tid);
  __syncthreads();

  const int lane = tid & 31, wv = tid >> 5;
  const int wM = wv & 3, wN = wv >> 2;
  const int lN = lane & 15, lM = lane & 15, lH = lane >> 4;

  v8f acc0 = {}, acc1 = {};
  const _Float16* Brow = W2h + (size_t)i * CC * CC + (size_t)(ch0 + wN * 32 + lN) * CC + 16 * lH;
  const _Float16* Abp  = xT + (wM * 16 + lM) * LDS_STRIDE + 16 * lH;
  #pragma unroll
  for (int k0 = 0; k0 < CC; k0 += KSTEP) {
    v16h b0 = *(const v16h*)(Brow + k0);
    v16h b1 = *(const v16h*)(Brow + 16 * CC + k0);
    v16h a  = combine16(*(const v8h*)(Abp + k0), *(const v8h*)(Abp + k0 + 8));
    acc0 = __builtin_amdgcn_wmma_f32_16x16x32_f16(false, a, false, b0, (short)0, acc0, false, false);
    acc1 = __builtin_amdgcn_wmma_f32_16x16x32_f16(false, a, false, b1, (short)0, acc1, false, false);
  }
  #pragma unroll
  for (int f = 0; f < 2; ++f) {
    int o = ch0 + wN * 32 + f * 16 + lN;
    float bv = cbias[i * CC + o];
    float* dst = cbuf + ((size_t)(i * BB + bI) * CC + o) * HW + pix0 + wM * 16 + 8 * lH;
    v8f A = f ? acc1 : acc0;
    #pragma unroll
    for (int r = 0; r < 8; ++r)
      __builtin_nontemporal_store(A[r] + bv, dst + r);
  }
}

// Fused: per (b, 64-pixel, 64-channel) tile compute all 25 targets, threshold,
// softmax over j, weighted sum -> out[i][b][o][hw].  Weight tiles arrive via
// TDM (tensor_load_to_lds) double-buffered across the i loop.
__global__ void fused_attn(const _Float16* __restrict__ Ah, const _Float16* __restrict__ xh,
                           const float* __restrict__ cbuf, float* __restrict__ out) {
  extern __shared__ _Float16 smem[];   // 5*TILE_J x-tiles + 2*WTILE weight buffers
  const int tid = threadIdx.x;
  const int bI = blockIdx.y;
  const int pix0 = (blockIdx.x & 15) * MT;
  const int ch0  = (blockIdx.x >> 4) * NT;
  const int lane = tid & 31, wv = tid >> 5;
  const int wM = wv & 3, wN = wv >> 2;
  const int lN = lane & 15, lM = lane & 15, lH = lane >> 4;

  // kick off TDM for i=0's weight tile; it runs while we stage x tiles
  if (wv == 0)
    tdm_load_weights(Ah + (size_t)ch0 * CC, XTILES_BYTES);

  for (int j = 0; j < NN; ++j)
    stage_xT(xh + ((size_t)(j * BB + bI) * CC) * HW + pix0, smem + j * TILE_J, tid);
  __syncthreads();

  const _Float16* Abase = smem + (wM * 16 + lM) * LDS_STRIDE + 16 * lH;

  for (int i = 0; i < NN; ++i) {
    if (wv == 0) {
      if (i + 1 < NN) {   // prefetch next weight tile into the other buffer
        tdm_load_weights(Ah + (size_t)(i + 1) * CC * CC + (size_t)ch0 * CC,
                         XTILES_BYTES + (uint32_t)(((i + 1) & 1) * WTILE * 2));
        __builtin_amdgcn_s_wait_tensorcnt(1);   // tile i is resident
      } else {
        __builtin_amdgcn_s_wait_tensorcnt(0);
      }
    }
    __syncthreads();   // weight tile i visible to all waves

    const _Float16* wT = smem + NN * TILE_J + (i & 1) * WTILE;
    const _Float16* Brow = wT + (size_t)(wN * 32 + lN) * LDS_STRIDE + 16 * lH;

    v8f acc[NN][2] = {};
    for (int k0 = 0; k0 < CC; k0 += KSTEP) {
      v16h b0 = *(const v16h*)(Brow + k0);
      v16h b1 = *(const v16h*)(Brow + 16 * LDS_STRIDE + k0);
      #pragma unroll
      for (int j = 0; j < NN; ++j) {
        v16h a = combine16(*(const v8h*)(Abase + j * TILE_J + k0),
                           *(const v8h*)(Abase + j * TILE_J + k0 + 8));
        acc[j][0] = __builtin_amdgcn_wmma_f32_16x16x32_f16(false, a, false, b0, (short)0, acc[j][0], false, false);
        acc[j][1] = __builtin_amdgcn_wmma_f32_16x16x32_f16(false, a, false, b1, (short)0, acc[j][1], false, false);
      }
    }
    // epilogue: dist -> threshold -> softmax over j -> weighted sum of x_j
    #pragma unroll
    for (int f = 0; f < 2; ++f) {
      const int o = ch0 + wN * 32 + f * 16 + lN;
      const size_t rowoff = ((size_t)(i * BB + bI) * CC + o) * HW + pix0 + wM * 16 + 8 * lH;
      const float* cb = cbuf + rowoff;
      float* dst = out + rowoff;
      #pragma unroll
      for (int r = 0; r < 8; ++r) {
        const float cadd = __builtin_nontemporal_load(cb + r);
        const int mloc = wM * 16 + 8 * lH + r;
        float e[NN], xv[NN];
        #pragma unroll
        for (int j = 0; j < NN; ++j) {
          float t  = acc[j][f][r] + cadd;
          float xj = (float)smem[j * TILE_J + mloc * LDS_STRIDE + o];
          float d  = fabsf(xj - t);
          e[j]  = d > THRESH ? d : 0.0f;
          xv[j] = xj;
        }
        float mx = e[0];
        #pragma unroll
        for (int j = 1; j < NN; ++j) mx = fmaxf(mx, e[j]);
        float s = 0.0f, hs = 0.0f;
        #pragma unroll
        for (int j = 0; j < NN; ++j) {
          float wgt = __expf(e[j] - mx);
          s += wgt; hs += wgt * xv[j];
        }
        __builtin_nontemporal_store(hs / s, dst + r);
      }
    }
    __syncthreads();   // everyone done with weight buffer (i&1) before it is overwritten
  }
}

extern "C" void kernel_launch(void* const* d_in, const int* in_sizes, int n_in,
                              void* d_out, int out_size, void* d_ws, size_t ws_size,
                              hipStream_t stream) {
  const float* x  = (const float*)d_in[0];   // [5,8,256,32,32]
  const float* w  = (const float*)d_in[1];   // [5,256,512]
  const float* cb = (const float*)d_in[2];   // [5,256]
  float* out = (float*)d_out;

  char* ws = (char*)d_ws;
  _Float16* Ah   = (_Float16*)(ws + WS_AH);
  _Float16* W2h  = (_Float16*)(ws + WS_W2H);
  _Float16* xh   = (_Float16*)(ws + WS_XH);
  float*    cbuf = (float*)(ws + WS_CBUF);

  prep_weights<<<(NN * CC * CC + 255) / 256, 256, 0, stream>>>(w, Ah, W2h);
  prep_x<<<(NN * BB * CC * HW) / (4 * 256), 256, 0, stream>>>(x, xh);

  dim3 gci(64, BB, NN);
  ci_gemm<<<gci, 256, 0, stream>>>(W2h, xh, cb, cbuf);

  const int smemBytes = (NN * TILE_J + 2 * WTILE) * (int)sizeof(_Float16); // 236544 B
  (void)hipFuncSetAttribute((const void*)fused_attn,
                            hipFuncAttributeMaxDynamicSharedMemorySize, smemBytes);
  dim3 gm(64, BB);
  fused_attn<<<gm, 256, smemBytes, stream>>>(Ah, xh, cbuf, out);
}
